// PatchTTCN_Mamba_Encoder_15178414424116
// MI455X (gfx1250) — matrix operand order repaired and checked by reading, not compiled, not run on hardware
//
#include <hip/hip_runtime.h>
#include <math.h>

#define B_ 64
#define S_ 4096
#define D_ 8
#define NPATCH_ 128
#define PLEN_ 32
#define HID_ 128
#define TTCN_ 127
#define NEG_ (-1e8f)

typedef __bf16 bf16;
typedef bf16 v16bf __attribute__((ext_vector_type(16)));
typedef bf16 v8bf  __attribute__((ext_vector_type(8)));
typedef float v8f  __attribute__((ext_vector_type(8)));

__device__ __forceinline__ v16bf cat8(v8bf lo, v8bf hi) {
    return __builtin_shufflevector(lo, hi, 0,1,2,3,4,5,6,7,8,9,10,11,12,13,14,15);
}

// A-fragment: 16-bit A layout -> per lane K pattern {k0..k0+7, k0+16..k0+23}
__device__ __forceinline__ v16bf load_frag_rowpair(const bf16* p) {
    v8bf lo = *(const v8bf*)(p);
    v8bf hi = *(const v8bf*)(p + 16);
    return cat8(lo, hi);
}
// B-fragment: 16 contiguous K values starting at kbase + 16*half
__device__ __forceinline__ v16bf load_frag_contig(const bf16* p) {
    v8bf lo = *(const v8bf*)(p);
    v8bf hi = *(const v8bf*)(p + 8);
    return cat8(lo, hi);
}

__device__ __forceinline__ v8f wmma_bf16(v16bf a, v16bf b, v8f c) {
    return __builtin_amdgcn_wmma_f32_16x16x32_bf16(false, a, false, b, (short)0, c, false, false);
}

// ---------------------------------------------------------------------------
// Kernel 0: weight prep — convert to bf16, transpose to N-major, zero-pad.
// W1t[128][32], W2t[128][128], W3t[1024][128]
// ---------------------------------------------------------------------------
__global__ void prep_kernel(const float* __restrict__ W1, const float* __restrict__ W2,
                            const float* __restrict__ W3,
                            bf16* __restrict__ W1t, bf16* __restrict__ W2t,
                            bf16* __restrict__ W3t) {
    int i = blockIdx.x * blockDim.x + threadIdx.x;
    if (i < 128 * 32) {
        int n = i / 32, k = i % 32;
        float v = (n < TTCN_ && k < D_) ? W1[k * TTCN_ + n] : 0.f;
        W1t[i] = (bf16)v;
    }
    int j = i - 128 * 32;
    if (j >= 0 && j < 128 * 128) {
        int n = j / 128, k = j % 128;
        float v = (n < TTCN_ && k < TTCN_) ? W2[k * TTCN_ + n] : 0.f;
        W2t[j] = (bf16)v;
    }
    int l = i - 128 * 32 - 128 * 128;
    if (l >= 0 && l < 1024 * 128) {
        int n = l / 128, k = l % 128;
        float v = (n < 1016 && k < TTCN_) ? W3[k * 1016 + n] : 0.f;
        W3t[l] = (bf16)v;
    }
}

// ---------------------------------------------------------------------------
// Kernel 1: patchify. One block per batch row. Sorted indices -> slot via
// lower_bound, exactly matching the reference cumsum slot assignment.
// ---------------------------------------------------------------------------
__global__ __launch_bounds__(256) void patchify_kernel(const float* __restrict__ mi,
                                                       const int* __restrict__ pidx,
                                                       float* __restrict__ Xp,
                                                       unsigned* __restrict__ maskw) {
    __shared__ int sidx[S_];
    __shared__ unsigned smask[NPATCH_];
    const int b = blockIdx.x;
    const int tid = threadIdx.x;
    for (int s = tid; s < S_; s += 256) sidx[s] = pidx[b * S_ + s];
    for (int i = tid; i < NPATCH_; i += 256) smask[i] = 0u;
    float4* xz = (float4*)(Xp + (size_t)b * NPATCH_ * PLEN_ * D_);
    for (int i = tid; i < NPATCH_ * PLEN_ * D_ / 4; i += 256) xz[i] = make_float4(0.f, 0.f, 0.f, 0.f);
    __syncthreads();
    for (int s = tid; s < S_; s += 256) {
        int pv = sidx[s];
        if (pv > 0 && pv <= NPATCH_) {
            int lo = 0, hi = S_;
            while (lo < hi) { int mid = (lo + hi) >> 1; if (sidx[mid] < pv) lo = mid + 1; else hi = mid; }
            int slot = s - lo;            // rank within the (sorted) run of equal ids
            if (slot < PLEN_) {
                int pos = pv - 1;
                const float4* src = (const float4*)(mi + ((size_t)b * S_ + s) * D_);
                float4* dst = (float4*)(Xp + (((size_t)b * NPATCH_ + pos) * PLEN_ + slot) * D_);
                dst[0] = src[0];
                dst[1] = src[1];
                atomicOr(&smask[pos], 1u << slot);
            }
        }
    }
    __syncthreads();
    for (int i = tid; i < NPATCH_; i += 256) maskw[b * NPATCH_ + i] = smask[i];
}

// ---------------------------------------------------------------------------
// Kernel 2: per-patch TTCN. One wave per patch (4 waves/block).
// ---------------------------------------------------------------------------
__global__ __launch_bounds__(128) void patch_kernel(const float* __restrict__ Xp,
                                                    const unsigned* __restrict__ maskw,
                                                    const bf16* __restrict__ W1t,
                                                    const bf16* __restrict__ W2t,
                                                    const bf16* __restrict__ W3t,
                                                    const float* __restrict__ b1,
                                                    const float* __restrict__ b2,
                                                    const float* __restrict__ b3,
                                                    const float* __restrict__ Tb,
                                                    float* __restrict__ xraw) {
    __shared__ __align__(16) bf16 xpbf[4][32 * 32];
    __shared__ __align__(16) bf16 hbuf[4][32 * 128];
    __shared__ float ttsm[4][128];

    const int wid = threadIdx.x >> 5;
    const int lane = threadIdx.x & 31;
    const int bp = blockIdx.x * 4 + wid;   // patch id 0..8191
    const int lm = lane & 15;
    const int half = lane >> 4;

    bf16* xq = xpbf[wid];
    bf16* h = hbuf[wid];
    float* ttw = ttsm[wid];

    const float* xpg = Xp + (size_t)bp * (PLEN_ * D_);
    const unsigned mbits = maskw[bp];

    // stage Xp -> bf16 (32x32, cols >= D zero-padded)
    for (int i = lane; i < 32 * 32; i += 32) {
        int r = i >> 5, k = i & 31;
        float v = (k < D_) ? xpg[r * D_ + k] : 0.f;
        xq[i] = (bf16)v;
    }
    // per-lane Xp column registers: p = lane&7 is invariant across all N-tiles
    const int p = lane & 7;
    float xpr[16];
#pragma unroll
    for (int mt = 0; mt < 2; ++mt)
#pragma unroll
        for (int j = 0; j < 8; ++j) {
            int r = 16 * mt + 8 * half + j;
            xpr[mt * 8 + j] = xpg[r * D_ + p];
        }

    // ---- GEMM1: h1 = relu(Xp @ W1 + b1)  (M=32, K=32pad, N=128) ----
    {
        v16bf a0 = load_frag_rowpair(xq + lm * 32 + 8 * half);
        v16bf a1 = load_frag_rowpair(xq + (lm + 16) * 32 + 8 * half);
#pragma unroll
        for (int nt = 0; nt < 8; ++nt) {
            int n = nt * 16 + lm;
            v16bf bfr = load_frag_rowpair(W1t + n * 32 + 8 * half);
            v8f c0 = {}; v8f c1 = {};
            c0 = wmma_bf16(a0, bfr, c0);
            c1 = wmma_bf16(a1, bfr, c1);
            float bias = (n < TTCN_) ? b1[n] : 0.f;
#pragma unroll
            for (int j = 0; j < 8; ++j) {
                int r0 = 8 * half + j;
                float v0 = c0[j] + bias; v0 = v0 > 0.f ? v0 : 0.f;
                float v1 = c1[j] + bias; v1 = v1 > 0.f ? v1 : 0.f;
                h[r0 * 128 + n] = (bf16)v0;
                h[(16 + r0) * 128 + n] = (bf16)v1;
            }
        }
    }

    // ---- GEMM2: h2 = relu(h1 @ W2 + b2)  (M=32, K=128, N=128) ----
    v16bf af[2][4];
#pragma unroll
    for (int mt = 0; mt < 2; ++mt)
#pragma unroll
        for (int kc = 0; kc < 4; ++kc)
            af[mt][kc] = load_frag_rowpair(h + (lm + 16 * mt) * 128 + 32 * kc + 8 * half);
#pragma unroll
    for (int nt = 0; nt < 8; ++nt) {
        int n = nt * 16 + lm;
        v8f c0 = {}; v8f c1 = {};
#pragma unroll
        for (int kc = 0; kc < 4; ++kc) {
            v16bf bfr = load_frag_rowpair(W2t + n * 128 + 32 * kc + 8 * half);
            c0 = wmma_bf16(af[0][kc], bfr, c0);
            c1 = wmma_bf16(af[1][kc], bfr, c1);
        }
        float bias = (n < TTCN_) ? b2[n] : 0.f;
#pragma unroll
        for (int j = 0; j < 8; ++j) {
            int r0 = 8 * half + j;
            float v0 = c0[j] + bias; v0 = v0 > 0.f ? v0 : 0.f;
            float v1 = c1[j] + bias; v1 = v1 > 0.f ? v1 : 0.f;
            h[r0 * 128 + n] = (bf16)v0;       // overwrite h1 (already in regs)
            h[(16 + r0) * 128 + n] = (bf16)v1;
        }
    }

    // preload h2 A-fragments (reused across all 64 N-tiles of the big GEMM)
#pragma unroll
    for (int mt = 0; mt < 2; ++mt)
#pragma unroll
        for (int kc = 0; kc < 4; ++kc)
            af[mt][kc] = load_frag_rowpair(h + (lm + 16 * mt) * 128 + 32 * kc + 8 * half);

    // ---- GEMM3 + fused mask/softmax/einsum  (N = 1024, tiles of 16) ----
    for (int tile = 0; tile < 64; ++tile) {
        int n = tile * 16 + lm;               // column = t*8 + p
        // prefetch next tile's W3 row into cache (global_prefetch_b8)
        if (tile + 1 < 64) {
            __builtin_prefetch(W3t + (n + 16) * 128 + 32 * 0 + 16 * half, 0, 3);
            __builtin_prefetch(W3t + (n + 16) * 128 + 32 * 2 + 16 * half, 0, 3);
        }
        float bias = (n < 1016) ? b3[n] : 0.f;
        v8f c0 = {}; v8f c1 = {};
#pragma unroll
        for (int kc = 0; kc < 4; ++kc) {
            v16bf bfr = load_frag_rowpair(W3t + n * 128 + 32 * kc + 8 * half);
            c0 = wmma_bf16(af[0][kc], bfr, c0);
            c1 = wmma_bf16(af[1][kc], bfr, c1);
        }
        // lane holds rows {8*half+j} and {16+8*half+j} of its column
        float fv[16], e[16];
        float mx = NEG_;
#pragma unroll
        for (int mt = 0; mt < 2; ++mt)
#pragma unroll
            for (int j = 0; j < 8; ++j) {
                int r = 16 * mt + 8 * half + j;
                float x = (mt ? c1[j] : c0[j]) + bias;
                x = ((mbits >> r) & 1u) ? x : NEG_;
                fv[mt * 8 + j] = x;
                mx = fmaxf(mx, x);
            }
        mx = fmaxf(mx, __shfl_xor(mx, 16, 32));   // join the two row-halves
        float ssum = 0.f;
#pragma unroll
        for (int i = 0; i < 16; ++i) { e[i] = __expf(fv[i] - mx); ssum += e[i]; }
        ssum += __shfl_xor(ssum, 16, 32);
        float s = 0.f;
#pragma unroll
        for (int i = 0; i < 16; ++i) s += e[i] * xpr[i];
        s /= ssum;
        s += __shfl_xor(s, 16, 32);               // full column sum
        s += __shfl_xor(s, 4, 32);                // reduce the 8 p-columns per t
        s += __shfl_xor(s, 2, 32);
        s += __shfl_xor(s, 1, 32);
        if (lane == 0) ttw[2 * tile] = s;
        if (lane == 8 && (2 * tile + 1) < TTCN_) ttw[2 * tile + 1] = s;
    }

    // patch_feat = relu(ttcn + T_bias); channel 127 = patch_exists
    for (int t = lane; t < HID_; t += 32) {
        float v;
        if (t < TTCN_) { v = ttw[t] + Tb[t]; v = v > 0.f ? v : 0.f; }
        else v = mbits ? 1.f : 0.f;
        xraw[(size_t)bp * HID_ + t] = v;
    }
}

// ---------------------------------------------------------------------------
// Kernel 3: positional encoding + masked mean + last valid token.
// ---------------------------------------------------------------------------
__global__ __launch_bounds__(128) void finalize_kernel(const float* __restrict__ xraw,
                                                       const float* __restrict__ pmask,
                                                       float* __restrict__ out) {
    const int b = blockIdx.x, t = threadIdx.x;
    int j = t >> 1;
    float div = __expf(-(float)(2 * j) * 9.210340371976184f / (float)HID_);
    float sum = 0.f, den = 0.f, lastv = 0.f;
    for (int pp = 0; pp < NPATCH_; ++pp) {
        float pm = pmask[b * NPATCH_ + pp];
        float pe = (t & 1) ? __cosf((float)pp * div) : __sinf((float)pp * div);
        float xv = (xraw[((size_t)b * NPATCH_ + pp) * HID_ + t] * pm + pe) * pm;
        sum += xv * pm;
        den += pm;
        if (pm > 0.f) lastv = xv;
    }
    float denom = den > 1.f ? den : 1.f;
    out[b * (2 * HID_) + t] = lastv;
    out[b * (2 * HID_) + HID_ + t] = sum / denom;
}

extern "C" void kernel_launch(void* const* d_in, const int* in_sizes, int n_in,
                              void* d_out, int out_size, void* d_ws, size_t ws_size,
                              hipStream_t stream) {
    const float* mi   = (const float*)d_in[0];
    const int*   pidx = (const int*)d_in[1];
    const float* pm   = (const float*)d_in[2];
    const float* W1   = (const float*)d_in[3];
    const float* b1   = (const float*)d_in[4];
    const float* W2   = (const float*)d_in[5];
    const float* b2   = (const float*)d_in[6];
    const float* W3   = (const float*)d_in[7];
    const float* b3   = (const float*)d_in[8];
    const float* Tb   = (const float*)d_in[9];
    (void)in_sizes; (void)n_in; (void)out_size; (void)ws_size;

    char* ws = (char*)d_ws;
    float*    Xp    = (float*)(ws);                  // 8 MB  : [8192][32][8] f32
    unsigned* maskw = (unsigned*)(ws + 8388608);     // 32 KB : [8192] u32
    bf16*     W1t   = (bf16*)(ws + 8421376);         // 8 KB
    bf16*     W2t   = (bf16*)(ws + 8429568);         // 32 KB
    bf16*     W3t   = (bf16*)(ws + 8462336);         // 256 KB
    float*    xraw  = (float*)(ws + 8724480);        // 4 MB  : [8192][128] f32

    prep_kernel<<<592, 256, 0, stream>>>(W1, W2, W3, W1t, W2t, W3t);
    patchify_kernel<<<B_, 256, 0, stream>>>(mi, pidx, Xp, maskw);
    patch_kernel<<<2048, 128, 0, stream>>>(Xp, maskw, W1t, W2t, W3t, b1, b2, b3, Tb, xraw);
    finalize_kernel<<<B_, 128, 0, stream>>>(xraw, pm, (float*)d_out);
}
